// SimpleAttention_76536317215262
// MI455X (gfx1250) — compile-verified, matrix operand order checked by reading, not run on hardware
//
#include <hip/hip_runtime.h>
#include <math.h>

// ---------------------------------------------------------------------------
// SimpleAttention forward for MI455X (gfx1250, wave32, WMMA, async-LDS).
//
// The reference's RMSNorm + qkv GEMM is dead code (qkv is deleted); output
// depends only on q / cache_K / cache_V.  Single-pass flash attention:
//   scores = (q @ K^T)/sqrt(D); out = softmax(scores) @ V
// H=32, M=512, P=4096, D=128.  f16 WMMA (f32 accum), f32 softmax.
//
// Data movement: K/V chunks (f32) are streamed into a double-buffered raw LDS
// staging area with GLOBAL_LOAD_ASYNC_TO_LDS_B128 (ASYNCcnt), overlapping the
// next chunk's HBM fetch with the current chunk's WMMA work, then converted
// LDS->LDS to f16 fragment layouts (K row-major, V transposed).
// ---------------------------------------------------------------------------

typedef __attribute__((ext_vector_type(16))) _Float16 v16h;
typedef __attribute__((ext_vector_type(8)))  _Float16 v8h;
typedef __attribute__((ext_vector_type(4)))  _Float16 v4h;
typedef __attribute__((ext_vector_type(8)))  float    v8f;

#define HEADS   32
#define MROWS   512
#define DDIM    128
#define PLEN    4096
#define PC      32      // KV positions per pipeline step
#define NSTEPS  (PLEN / PC)
#define NWAVES  4       // waves per block; each wave owns 16 q rows
#define KPAD    16      // f16 pads: keep 16B alignment, spread LDS banks
#define VPAD    16
#define PPAD    16

__device__ __forceinline__ v16h combine8(const v8h lo, const v8h hi) {
    v16h b;
#pragma unroll
    for (int i = 0; i < 8; ++i) { b[i] = lo[i]; b[8 + i] = hi[i]; }
    return b;
}

// Low 32 bits of a generic LDS address are the LDS byte offset (ISA 10.2).
__device__ __forceinline__ uint32_t lds_off32(const void* p) {
    return (uint32_t)(uintptr_t)p;
}

// GLOBAL_LOAD_ASYNC_TO_LDS_B128, GVS mode: mem = SADDR64 + VADDR32 + IOFFSET,
// LDS dest = LDS_BASE + VDST32.  Tracked with ASYNCcnt.
__device__ __forceinline__ void async_ld_b128(uint32_t lds_byte, uint32_t g_byte,
                                              const void* base) {
    asm volatile("global_load_async_to_lds_b128 %0, %1, %2"
                 :: "v"(lds_byte), "v"(g_byte), "s"(base) : "memory");
}

__global__ __launch_bounds__(128)
void fa_fwd_wmma_kernel(const float* __restrict__ q,
                        const float* __restrict__ cache_K,
                        const float* __restrict__ cache_V,
                        float* __restrict__ out)
{
    // Raw f32 double buffers filled by the async engine (contiguous chunks).
    __shared__ __align__(16) float raw_k[2][PC * DDIM];   // 2 x 16 KB
    __shared__ __align__(16) float raw_v[2][PC * DDIM];   // 2 x 16 KB
    // f16 fragment buffers.
    __shared__ _Float16 lds_k[PC][DDIM + KPAD];           // [p][d]     9 KB
    __shared__ _Float16 lds_v[DDIM][PC + VPAD];           // [d][p]    12 KB
    __shared__ _Float16 lds_p[NWAVES][16][PC + PPAD];     // C->A tile  6 KB

    const int mtiles = MROWS / (NWAVES * 16);             // 8
    const int h    = blockIdx.x / mtiles;
    const int mt   = blockIdx.x % mtiles;
    const int tid  = threadIdx.x;
    const int wave = tid >> 5;
    const int lane = tid & 31;
    const int lrow = lane & 15;     // A-row / B-col / C-col index
    const int lh   = lane >> 4;     // lane half (selects K-group / M-group)

    const int   m0    = mt * (NWAVES * 16) + wave * 16;
    const float scale = 0.08838834764831845f;             // 1/sqrt(128)

    const float* Kh = cache_K + (size_t)h * PLEN * DDIM;
    const float* Vh = cache_V + (size_t)h * PLEN * DDIM;

    // ---- preload q as 4 f16 A-fragments (16 rows x 32 d each), pre-scaled --
    // 16-bit A layout: lanes 0-15 hold K = {0..7,16..23}, lanes 16-31 hold
    // K = {8..15,24..31} within each 32-wide d-chunk.
    const float* qrow = q + ((size_t)h * MROWS + m0 + lrow) * DDIM;
    v16h aq[4];
#pragma unroll
    for (int kc = 0; kc < 4; ++kc) {
        const float* qp = qrow + kc * 32 + lh * 8;
        float4 a0 = *(const float4*)(qp + 0);
        float4 a1 = *(const float4*)(qp + 4);
        float4 a2 = *(const float4*)(qp + 16);
        float4 a3 = *(const float4*)(qp + 20);
        v16h a;
        a[0]  = (_Float16)(a0.x * scale); a[1]  = (_Float16)(a0.y * scale);
        a[2]  = (_Float16)(a0.z * scale); a[3]  = (_Float16)(a0.w * scale);
        a[4]  = (_Float16)(a1.x * scale); a[5]  = (_Float16)(a1.y * scale);
        a[6]  = (_Float16)(a1.z * scale); a[7]  = (_Float16)(a1.w * scale);
        a[8]  = (_Float16)(a2.x * scale); a[9]  = (_Float16)(a2.y * scale);
        a[10] = (_Float16)(a2.z * scale); a[11] = (_Float16)(a2.w * scale);
        a[12] = (_Float16)(a3.x * scale); a[13] = (_Float16)(a3.y * scale);
        a[14] = (_Float16)(a3.z * scale); a[15] = (_Float16)(a3.w * scale);
        aq[kc] = a;
    }

    // ---- streaming-softmax state (per lane: 8 rows = r + 8*lane_half) ------
    float mrun[8], lrun[8];
    v8f   acc[8];                  // O accumulator: 8 d-tiles of C layout
#pragma unroll
    for (int r = 0; r < 8; ++r) { mrun[r] = -INFINITY; lrun[r] = 0.0f; }
    {
        v8f z = {};
#pragma unroll
        for (int dt = 0; dt < 8; ++dt) acc[dt] = z;
    }

    // ---- prologue: async-issue chunk 0 into buffer 0 -----------------------
    // Chunk is a contiguous 16 KB block of K_h (and of V_h): linear copy.
    // 128 threads x 8 b128 = 1024 x 16 B = 16 KB per matrix.
    {
        const float* Kc = Kh;
        const float* Vc = Vh;
#pragma unroll
        for (int j = 0; j < 8; ++j) {
            const uint32_t byt = (uint32_t)(j * 128 + tid) * 16u;
            async_ld_b128(lds_off32(&raw_k[0][0]) + byt, byt, Kc);
            async_ld_b128(lds_off32(&raw_v[0][0]) + byt, byt, Vc);
        }
    }

    // ---- main loop over KV chunks ------------------------------------------
    for (int step = 0; step < NSTEPS; ++step) {
        const int buf = step & 1;

        if (step + 1 < NSTEPS) {
            // issue next chunk into the other buffer, then wait for current
            const float* Kc = Kh + (size_t)(step + 1) * PC * DDIM;
            const float* Vc = Vh + (size_t)(step + 1) * PC * DDIM;
            const uint32_t kb = lds_off32(&raw_k[buf ^ 1][0]);
            const uint32_t vb = lds_off32(&raw_v[buf ^ 1][0]);
#pragma unroll
            for (int j = 0; j < 8; ++j) {
                const uint32_t byt = (uint32_t)(j * 128 + tid) * 16u;
                async_ld_b128(kb + byt, byt, Kc);
                async_ld_b128(vb + byt, byt, Vc);
            }
            // 16 outstanding = the loads just issued; current chunk complete.
            asm volatile("s_wait_asynccnt 0x10" ::: "memory");
        } else {
            asm volatile("s_wait_asynccnt 0x0" ::: "memory");
        }
        __syncthreads();   // async data visible to all waves; prior compute done

        // ---- LDS->LDS convert: raw f32 -> f16 frag layouts -----------------
        {
            const int row = tid >> 2;          // 4 threads per p-row
            const int db  = (tid & 3) * 32;    // 32 d's per thread
            const float* rk = &raw_k[buf][row * DDIM];
            const float* rv = &raw_v[buf][row * DDIM];
#pragma unroll
            for (int j = 0; j < 8; ++j) {
                const int d = db + j * 4;
                float4 kv = *(const float4*)(rk + d);
                float4 vv = *(const float4*)(rv + d);
                v4h kh;
                kh[0] = (_Float16)kv.x; kh[1] = (_Float16)kv.y;
                kh[2] = (_Float16)kv.z; kh[3] = (_Float16)kv.w;
                *(v4h*)&lds_k[row][d] = kh;                 // row-major
                lds_v[d + 0][row] = (_Float16)vv.x;         // transposed
                lds_v[d + 1][row] = (_Float16)vv.y;
                lds_v[d + 2][row] = (_Float16)vv.z;
                lds_v[d + 3][row] = (_Float16)vv.w;
            }
        }
        __syncthreads();

        // ---- S = q @ K^T : batch all 8 B-fragments, then 8 WMMAs ----------
        v16h bk[8];
#pragma unroll
        for (int kc = 0; kc < 4; ++kc) {
            const int dof = kc * 32 + lh * 8;
            bk[kc * 2 + 0] = combine8(*(const v8h*)&lds_k[lrow][dof],
                                      *(const v8h*)&lds_k[lrow][dof + 16]);
            bk[kc * 2 + 1] = combine8(*(const v8h*)&lds_k[16 + lrow][dof],
                                      *(const v8h*)&lds_k[16 + lrow][dof + 16]);
        }
        v8f s0 = {}, s1 = {};
#pragma unroll
        for (int kc = 0; kc < 4; ++kc) {
            s0 = __builtin_amdgcn_wmma_f32_16x16x32_f16(
                     false, aq[kc], false, bk[kc * 2 + 0], (short)0, s0, false, false);
            s1 = __builtin_amdgcn_wmma_f32_16x16x32_f16(
                     false, aq[kc], false, bk[kc * 2 + 1], (short)0, s1, false, false);
        }

        // ---- streaming softmax update (f32) -------------------------------
        float fac[8];
#pragma unroll
        for (int r = 0; r < 8; ++r) {
            float v = fmaxf(s0[r], s1[r]);
#pragma unroll
            for (int mk = 1; mk < 16; mk <<= 1)
                v = fmaxf(v, __shfl_xor(v, mk, 32));     // row-max over 16 lanes
            const float mnew = fmaxf(mrun[r], v);
            fac[r]  = __expf(mrun[r] - mnew);            // first step: exp(-inf)=0
            mrun[r] = mnew;
            const float e0 = __expf(s0[r] - mnew);
            const float e1 = __expf(s1[r] - mnew);
            float ps = e0 + e1;
#pragma unroll
            for (int mk = 1; mk < 16; mk <<= 1)
                ps += __shfl_xor(ps, mk, 32);            // row-sum
            lrun[r] = lrun[r] * fac[r] + ps;
            // relayout C(row=VGPR,col=lane) -> A(row=lane,col=VGPR) via LDS
            lds_p[wave][8 * lh + r][lrow]      = (_Float16)e0;
            lds_p[wave][8 * lh + r][16 + lrow] = (_Float16)e1;
        }
#pragma unroll
        for (int dt = 0; dt < 8; ++dt)
#pragma unroll
            for (int r = 0; r < 8; ++r) acc[dt][r] *= fac[r];

        // ---- O += P @ V : batch 8 V fragments, then 8 WMMAs ---------------
        // (same-wave LDS ops are in-order: lds_p stores precede this load)
        v16h ap = combine8(*(const v8h*)&lds_p[wave][lrow][lh * 8],
                           *(const v8h*)&lds_p[wave][lrow][16 + lh * 8]);
        v16h bv[8];
#pragma unroll
        for (int dt = 0; dt < 8; ++dt) {
            const int d = dt * 16 + lrow;
            bv[dt] = combine8(*(const v8h*)&lds_v[d][lh * 8],
                              *(const v8h*)&lds_v[d][16 + lh * 8]);
        }
#pragma unroll
        for (int dt = 0; dt < 8; ++dt)
            acc[dt] = __builtin_amdgcn_wmma_f32_16x16x32_f16(
                          false, ap, false, bv[dt], (short)0, acc[dt], false, false);
    }

    // ---- epilogue: out[m, h*128 + d] = O / l ------------------------------
#pragma unroll
    for (int r = 0; r < 8; ++r) {
        const float inv  = 1.0f / lrun[r];
        const int   grow = m0 + 8 * lh + r;
        float* orow = out + (size_t)grow * (HEADS * DDIM) + (size_t)h * DDIM;
#pragma unroll
        for (int dt = 0; dt < 8; ++dt)
            orow[dt * 16 + lrow] = acc[dt][r] * inv;
    }
}

extern "C" void kernel_launch(void* const* d_in, const int* in_sizes, int n_in,
                              void* d_out, int out_size, void* d_ws, size_t ws_size,
                              hipStream_t stream) {
    // inputs (setup_inputs order): X, weight, W, q, cache_K, cache_V.
    // X/weight/W feed only the deleted qkv — not needed.
    const float* q  = (const float*)d_in[3];
    const float* cK = (const float*)d_in[4];
    const float* cV = (const float*)d_in[5];
    float* out = (float*)d_out;

    dim3 grid(HEADS * (MROWS / (NWAVES * 16)));  // 32 * 8 = 256 blocks
    dim3 block(NWAVES * 32);                     // 128 threads = 4 wave32
    hipLaunchKernelGGL(fa_fwd_wmma_kernel, grid, block, 0, stream,
                       q, cK, cV, out);
    (void)in_sizes; (void)n_in; (void)out_size; (void)d_ws; (void)ws_size;
}
